// GDN_41240275976741
// MI455X (gfx1250) — compile-verified, matrix operand order checked by reading, not run on hardware
//
#include <hip/hip_runtime.h>
#include <hip/hip_bf16.h>
#include <math.h>

#define NODE   100
#define BATCHSZ 64
#define DIM    128
#define IN_DIM   5
#define TOPK    30
#define INTER  256
#define NB     (BATCHSZ * NODE)   // 6400
#define BN_EPS 1e-5f
#define SLOPE  0.2f

typedef float    v2f  __attribute__((ext_vector_type(2)));
typedef float    v8f  __attribute__((ext_vector_type(8)));
typedef _Float16 v16h __attribute__((ext_vector_type(16)));

// ---------------------------------------------------------------------------
// 1) cosine similarity matrix of emb (100x100)
__global__ void cos_kernel(const float* __restrict__ emb, float* __restrict__ cosm) {
    int i = blockIdx.x;
    int j = threadIdx.x;
    if (j >= NODE) return;
    float dot = 0.f, ni = 0.f, nj = 0.f;
    for (int k = 0; k < DIM; ++k) {
        float a = emb[i * DIM + k];
        float b = emb[j * DIM + k];
        dot += a * b; ni += a * a; nj += b * b;
    }
    cosm[i * NODE + j] = dot * rsqrtf(ni) * rsqrtf(nj);
}

// 2) top-30 per row (destructive selection; matches jax top_k tie order)
__global__ void topk_kernel(float* __restrict__ cosm, int* __restrict__ idx) {
    int i = threadIdx.x;
    if (i >= NODE) return;
    for (int k = 0; k < TOPK; ++k) {
        float best = -1e30f; int bj = 0;
        for (int j = 0; j < NODE; ++j) {
            float v = cosm[i * NODE + j];
            if (v > best) { best = v; bj = j; }
        }
        idx[i * TOPK + k] = bj;
        cosm[i * NODE + bj] = -1e30f;
    }
}

// 3) h = x @ w_lin^T (K=5) and per-row attention scalars s_i, s_j
__global__ void lin_kernel(const float* __restrict__ x, const float* __restrict__ emb,
                           const float* __restrict__ w_lin,
                           const float* __restrict__ att_i, const float* __restrict__ att_j,
                           const float* __restrict__ att_em_i, const float* __restrict__ att_em_j,
                           float* __restrict__ h, float* __restrict__ s_i, float* __restrict__ s_j) {
    __shared__ float ri[DIM];
    __shared__ float rj[DIM];
    int r = blockIdx.x, c = threadIdx.x;
    float hv = 0.f;
    for (int k = 0; k < IN_DIM; ++k) hv += x[r * IN_DIM + k] * w_lin[c * IN_DIM + k];
    h[r * DIM + c] = hv;
    float e = emb[(r % NODE) * DIM + c];
    ri[c] = hv * att_i[c] + e * att_em_i[c];
    rj[c] = hv * att_j[c] + e * att_em_j[c];
    __syncthreads();
    for (int s = DIM / 2; s > 0; s >>= 1) {
        if (c < s) { ri[c] += ri[c + s]; rj[c] += rj[c + s]; }
        __syncthreads();
    }
    if (c == 0) { s_i[r] = ri[0]; s_j[r] = rj[0]; }
}

// 4) per-target leaky-relu + segment softmax (30 contiguous edges) + weighted scatter
__global__ void attn_kernel(const int* __restrict__ idx,
                            const float* __restrict__ s_i, const float* __restrict__ s_j,
                            const float* __restrict__ h, const float* __restrict__ g_bias,
                            float* __restrict__ agg) {
    __shared__ float attw[TOPK];
    __shared__ int   srcs[TOPK];
    int r = blockIdx.x;
    int b = r / NODE;
    int i = r % NODE;
    int t = threadIdx.x;
    if (t < TOPK) {
        int sj = idx[i * TOPK + t];
        srcs[t] = b * NODE + sj;
        float al = s_i[r] + s_j[b * NODE + sj];
        attw[t] = (al > 0.f) ? al : SLOPE * al;
    }
    __syncthreads();
    if (t == 0) {
        float m = -1e30f;
        for (int k = 0; k < TOPK; ++k) m = fmaxf(m, attw[k]);
        float s = 0.f;
        for (int k = 0; k < TOPK; ++k) { attw[k] = __expf(attw[k] - m); s += attw[k]; }
        float inv = 1.f / (s + 1e-16f);
        for (int k = 0; k < TOPK; ++k) attw[k] *= inv;
    }
    __syncthreads();
    float acc = 0.f;
    for (int k = 0; k < TOPK; ++k) acc += attw[k] * h[srcs[k] * DIM + t];
    agg[r * DIM + t] = acc + g_bias[t];
}

// 5) per-channel batch stats (biased variance), one block per channel
__global__ void stats_kernel(const float* __restrict__ src, int rows, int cols,
                             float* __restrict__ mu, float* __restrict__ var) {
    __shared__ float s1[256];
    __shared__ float s2[256];
    int c = blockIdx.x, t = threadIdx.x;
    float a = 0.f, b = 0.f;
    for (int r = t; r < rows; r += 256) {
        float v = src[r * cols + c];
        a += v; b += v * v;
    }
    s1[t] = a; s2[t] = b;
    __syncthreads();
    for (int s = 128; s > 0; s >>= 1) {
        if (t < s) { s1[t] += s1[t + s]; s2[t] += s2[t + s]; }
        __syncthreads();
    }
    if (t == 0) {
        float m = s1[0] / (float)rows;
        mu[c] = m;
        var[c] = s2[0] / (float)rows - m * m;
    }
}

// 6) in-place: bn1 + relu, then * emb broadcast  (agg -> x3)
__global__ void bn1mul_kernel(float* __restrict__ a, const float* __restrict__ emb,
                              const float* __restrict__ mu, const float* __restrict__ var,
                              const float* __restrict__ g, const float* __restrict__ bb) {
    int r = blockIdx.x, c = threadIdx.x;
    float v = a[r * DIM + c];
    float y = (v - mu[c]) * rsqrtf(var[c] + BN_EPS) * g[c] + bb[c];
    y = fmaxf(y, 0.f);
    a[r * DIM + c] = y * emb[(r % NODE) * DIM + c];
}

// 7) in-place: bn2 + relu  (x3 -> x3n)
__global__ void bn2_kernel(float* __restrict__ a,
                           const float* __restrict__ mu, const float* __restrict__ var,
                           const float* __restrict__ g, const float* __restrict__ bb) {
    int r = blockIdx.x, c = threadIdx.x;
    float v = a[r * DIM + c];
    float y = (v - mu[c]) * rsqrtf(var[c] + BN_EPS) * g[c] + bb[c];
    a[r * DIM + c] = fmaxf(y, 0.f);
}

// 8) WMMA GEMM: O[6400,256] = A[6400,128] @ w1^T + b1
//    one 16x16 tile per wave, 8 waves per 256-thread block, 800 blocks.
__global__ void gemm_wmma_kernel(const float* __restrict__ A, const float* __restrict__ w1,
                                 const float* __restrict__ b1, float* __restrict__ O) {
    int wave = threadIdx.x >> 5;
    int lane = threadIdx.x & 31;
    int tile = blockIdx.x * 8 + wave;
    int m0 = (tile >> 4) << 4;     // 400 M tiles
    int n0 = (tile & 15) << 4;     // 16  N tiles
    int half = lane >> 4;          // K-pair selector
    int l = lane & 15;
    v8f c = {0.f, 0.f, 0.f, 0.f, 0.f, 0.f, 0.f, 0.f};
#if __has_builtin(__builtin_amdgcn_wmma_f32_16x16x4_f32)
    const float* arow = A  + (m0 + l) * DIM + 2 * half;   // A[m0+l][k..]
    const float* brow = w1 + (n0 + l) * DIM + 2 * half;   // B[k][n0+l] = w1[n0+l][k]
#pragma unroll
    for (int kk = 0; kk < DIM; kk += 4) {
        v2f af; af.x = arow[kk]; af.y = arow[kk + 1];
        v2f bf; bf.x = brow[kk]; bf.y = brow[kk + 1];
        c = __builtin_amdgcn_wmma_f32_16x16x4_f32(false, af, false, bf,
                                                  (short)0, c, false, false);
    }
#else
    // fallback: codegen-confirmed f16 path, K=32 per WMMA, on-the-fly convert
    const float* arow = A  + (m0 + l) * DIM;
    const float* brow = w1 + (n0 + l) * DIM;
    for (int kk = 0; kk < DIM; kk += 32) {
        v16h af, bf;
#pragma unroll
        for (int e = 0; e < 16; ++e) {
            int v = e >> 1, pos = e & 1;
            int K = (v < 4 ? v * 2 + half * 8 : 16 + (v - 4) * 2 + half * 8) + pos;
            af[e] = (_Float16)arow[kk + K];
            bf[e] = (_Float16)brow[kk + K];
        }
        c = __builtin_amdgcn_wmma_f32_16x16x32_f16(false, af, false, bf,
                                                   (short)0, c, false, false);
    }
#endif
#pragma unroll
    for (int v = 0; v < 8; ++v) {
        int row = m0 + v + 8 * half;   // C/D layout: vgpr v -> M=v (lanes<16) / v+8
        int col = n0 + l;
        O[row * INTER + col] = c[v] + b1[col];
    }
}

// 9) bn3 + relu + dot with w2 + b2 -> out[6400]
__global__ void final_kernel(const float* __restrict__ O,
                             const float* __restrict__ mu, const float* __restrict__ var,
                             const float* __restrict__ g, const float* __restrict__ bb,
                             const float* __restrict__ w2, const float* __restrict__ b2,
                             float* __restrict__ out) {
    __shared__ float s1[INTER];
    int r = blockIdx.x, t = threadIdx.x;
    float v = O[r * INTER + t];
    float y = (v - mu[t]) * rsqrtf(var[t] + BN_EPS) * g[t] + bb[t];
    y = fmaxf(y, 0.f);
    s1[t] = y * w2[t];
    __syncthreads();
    for (int s = INTER / 2; s > 0; s >>= 1) {
        if (t < s) s1[t] += s1[t + s];
        __syncthreads();
    }
    if (t == 0) out[r] = s1[0] + b2[0];
}

// ---------------------------------------------------------------------------
extern "C" void kernel_launch(void* const* d_in, const int* in_sizes, int n_in,
                              void* d_out, int out_size, void* d_ws, size_t ws_size,
                              hipStream_t stream) {
    const float* x        = (const float*)d_in[0];
    const float* emb      = (const float*)d_in[1];
    const float* w_lin    = (const float*)d_in[2];
    const float* att_i    = (const float*)d_in[3];
    const float* att_j    = (const float*)d_in[4];
    const float* att_em_i = (const float*)d_in[5];
    const float* att_em_j = (const float*)d_in[6];
    const float* g_bias   = (const float*)d_in[7];
    const float* bn1_g    = (const float*)d_in[8];
    const float* bn1_b    = (const float*)d_in[9];
    const float* bn2_g    = (const float*)d_in[10];
    const float* bn2_b    = (const float*)d_in[11];
    const float* bn3_g    = (const float*)d_in[12];
    const float* bn3_b    = (const float*)d_in[13];
    const float* w1       = (const float*)d_in[14];
    const float* b1       = (const float*)d_in[15];
    const float* w2       = (const float*)d_in[16];
    const float* b2       = (const float*)d_in[17];
    float* out = (float*)d_out;
    float* ws  = (float*)d_ws;

    // workspace layout (floats); everything is (re)written every call
    float* cosm = ws;                          // 10000
    int*   idx  = (int*)(ws + 10016);          // 3000
    float* h    = ws + 13024;                  // 819200
    float* s_i  = h + (size_t)NB * DIM;        // 6400
    float* s_j  = s_i + NB;                    // 6400
    float* agg  = s_j + NB;                    // 819200 (reused as x3 / x3n)
    float* o    = agg + (size_t)NB * DIM;      // 1638400
    float* st   = o + (size_t)NB * INTER;      // stats
    float* mu1 = st;        float* var1 = st + 128;
    float* mu2 = st + 256;  float* var2 = st + 384;
    float* mu3 = st + 512;  float* var3 = st + 768;

    cos_kernel<<<NODE, 128, 0, stream>>>(emb, cosm);
    topk_kernel<<<1, 128, 0, stream>>>(cosm, idx);
    lin_kernel<<<NB, DIM, 0, stream>>>(x, emb, w_lin, att_i, att_j, att_em_i, att_em_j,
                                       h, s_i, s_j);
    attn_kernel<<<NB, DIM, 0, stream>>>(idx, s_i, s_j, h, g_bias, agg);
    stats_kernel<<<DIM, 256, 0, stream>>>(agg, NB, DIM, mu1, var1);
    bn1mul_kernel<<<NB, DIM, 0, stream>>>(agg, emb, mu1, var1, bn1_g, bn1_b);
    stats_kernel<<<DIM, 256, 0, stream>>>(agg, NB, DIM, mu2, var2);
    bn2_kernel<<<NB, DIM, 0, stream>>>(agg, mu2, var2, bn2_g, bn2_b);
    gemm_wmma_kernel<<<NB / 8, 256, 0, stream>>>(agg, w1, b1, o);
    stats_kernel<<<INTER, 256, 0, stream>>>(o, NB, INTER, mu3, var3);
    final_kernel<<<NB, INTER, 0, stream>>>(o, mu3, var3, bn3_g, bn3_b, w2, b2, out);
}
